// MLP_4415226380429
// MI455X (gfx1250) — matrix operand order repaired and checked by reading, not compiled, not run on hardware
//
#include <hip/hip_runtime.h>
#include <hip/hip_bf16.h>

// ---------------- problem constants ----------------
#define R_RAYS 4096
#define S_SAMP 256
#define NPTS   (R_RAYS * S_SAMP)
#define GRID   400
#define APPD   27
#define FEAT   128

#define LOX -1.5f
#define LOY -1.67f
#define LOZ -1.0f
#define HIX 1.5f
#define HIY 1.67f
#define HIZ 1.0f
#define INVX (2.0f / (HIX - LOX))
#define INVY (2.0f / (HIY - LOY))
#define INVZ (2.0f / (HIZ - LOZ))
#define DIST_SCALE 25.0f
#define W_THRESH 0.0001f

// padded LDS row strides (in halves) to de-conflict 64-bank LDS
#define LDX 40    // X rows: 32 used (27 app + 3 viewdir + pad) + 8 pad
#define LDH 136   // H rows: 128 used + 8 pad
#define LDF 104   // feats rows: 96 used (72 feats + 24 zero-pad) + 8 pad

typedef __attribute__((ext_vector_type(16))) _Float16 v16h;
typedef __attribute__((ext_vector_type(8)))  float    v8f;
typedef __attribute__((ext_vector_type(4)))  unsigned int u32x4;
typedef __attribute__((ext_vector_type(8)))  int      i32x8;
typedef __attribute__((ext_vector_type(4)))  int      i32x4;

// ---------------- sampling helpers ----------------
__device__ __forceinline__ void bilin_setup(float x, float y, int& pidx, float& wx, float& wy) {
  float fx = (x + 1.0f) * (0.5f * (float)(GRID - 1));
  float fy = (y + 1.0f) * (0.5f * (float)(GRID - 1));
  int x0 = (int)floorf(fx); x0 = x0 < 0 ? 0 : (x0 > GRID - 2 ? GRID - 2 : x0);
  int y0 = (int)floorf(fy); y0 = y0 < 0 ? 0 : (y0 > GRID - 2 ? GRID - 2 : y0);
  wx = fminf(fmaxf(fx - (float)x0, 0.0f), 1.0f);
  wy = fminf(fmaxf(fy - (float)y0, 0.0f), 1.0f);
  pidx = y0 * GRID + x0;
}

__device__ __forceinline__ void lin_setup(float z, int& z0, float& wz) {
  float fz = (z + 1.0f) * (0.5f * (float)(GRID - 1));
  int zi = (int)floorf(fz); z0 = zi < 0 ? 0 : (zi > GRID - 2 ? GRID - 2 : zi);
  wz = fminf(fmaxf(fz - (float)z0, 0.0f), 1.0f);
}

// sum_c plane[c](x,y) * line[c](z)   (density path)
__device__ __forceinline__ float plane_line_sum(const float* __restrict__ plane,
                                                const float* __restrict__ line,
                                                int C, float x, float y, float z) {
  int pidx; float wx, wy; bilin_setup(x, y, pidx, wx, wy);
  int z0;   float wz;     lin_setup(z, z0, wz);
  float w00 = (1.0f - wx) * (1.0f - wy);
  float w01 = wx * (1.0f - wy);
  float w10 = (1.0f - wx) * wy;
  float w11 = wx * wy;
  float s = 0.0f;
  for (int c = 0; c < C; ++c) {
    const float* pp = plane + c * (GRID * GRID) + pidx;
    float pv = pp[0] * w00 + pp[1] * w01 + pp[GRID] * w10 + pp[GRID + 1] * w11;
    const float* lp = line + c * GRID + z0;
    float lv = lp[0] * (1.0f - wz) + lp[1] * wz;
    s += pv * lv;
  }
  return s;
}

// appearance path: store per-channel plane*line into the f16 feats row
__device__ __forceinline__ void feat_store(const float* __restrict__ plane,
                                           const float* __restrict__ line,
                                           int cstart, int ccount, int coff,
                                           float x, float y, float z,
                                           _Float16* __restrict__ frow) {
  int pidx; float wx, wy; bilin_setup(x, y, pidx, wx, wy);
  int z0;   float wz;     lin_setup(z, z0, wz);
  float w00 = (1.0f - wx) * (1.0f - wy);
  float w01 = wx * (1.0f - wy);
  float w10 = (1.0f - wx) * wy;
  float w11 = wx * wy;
  for (int c = cstart; c < cstart + ccount; ++c) {
    const float* pp = plane + c * (GRID * GRID) + pidx;
    float pv = pp[0] * w00 + pp[1] * w01 + pp[GRID] * w10 + pp[GRID + 1] * w11;
    const float* lp = line + c * GRID + z0;
    float lv = lp[0] * (1.0f - wz) + lp[1] * wz;
    frow[coff + c] = (_Float16)(pv * lv);
  }
}

// load one 16x32 f16 A-fragment from a padded LDS row buffer
__device__ __forceinline__ v16h load_afrag(const _Float16* __restrict__ row, int kbase) {
  v16h a;
#pragma unroll
  for (int j = 0; j < 8; ++j) a[j] = row[kbase + j];
#pragma unroll
  for (int j = 0; j < 8; ++j) a[8 + j] = row[kbase + 8 + j];
  return a;
}

// ---------------- K1: density -> sigma ----------------
__global__ void sigma_kernel(const float* __restrict__ pts,
                             const float* __restrict__ dp0, const float* __restrict__ dp1,
                             const float* __restrict__ dp2,
                             const float* __restrict__ dl0, const float* __restrict__ dl1,
                             const float* __restrict__ dl2,
                             float* __restrict__ sigma_out) {
  int p = blockIdx.x * blockDim.x + threadIdx.x;
  if (p >= NPTS) return;
  float px = pts[p * 3 + 0], py = pts[p * 3 + 1], pz = pts[p * 3 + 2];
  bool inb = !(px < LOX || px > HIX || py < LOY || py > HIY || pz < LOZ || pz > HIZ);
  float xn = (px - LOX) * INVX - 1.0f;
  float yn = (py - LOY) * INVY - 1.0f;
  float zn = (pz - LOZ) * INVZ - 1.0f;
  float s = 0.0f;
  s += plane_line_sum(dp0, dl0, 16, xn, yn, zn);  // (x,y) plane, z line
  s += plane_line_sum(dp1, dl1, 4,  xn, zn, yn);  // (x,z) plane, y line
  s += plane_line_sum(dp2, dl2, 4,  yn, zn, xn);  // (y,z) plane, x line
  s = fmaxf(s, 0.0f);
  sigma_out[p] = inb ? s : 0.0f;
}

// ---------------- K2: per-ray transmittance scan ----------------
__global__ void scan_kernel(const float* __restrict__ z_vals,
                            const float* __restrict__ sigma,
                            float* __restrict__ weight) {
  int r = blockIdx.x * blockDim.x + threadIdx.x;
  if (r >= R_RAYS) return;
  const float* z  = z_vals + (size_t)r * S_SAMP;
  const float* sg = sigma  + (size_t)r * S_SAMP;
  float* w        = weight + (size_t)r * S_SAMP;
  float T = 1.0f;
  for (int s = 0; s < S_SAMP; ++s) {
    float dist  = (s + 1 < S_SAMP) ? (z[s + 1] - z[s]) : 0.0f;
    float alpha = 1.0f - __expf(-sg[s] * dist * DIST_SCALE);
    w[s] = alpha * T;
    T *= (1.0f - alpha + 1e-10f);
  }
}

// ---------------- K3: appearance + WMMA einsum + WMMA MLP ----------------
__global__ __launch_bounds__(256) void mlp_kernel(
    const float* __restrict__ pts, const float* __restrict__ view_dirs,
    const float* __restrict__ ap0, const float* __restrict__ ap1, const float* __restrict__ ap2,
    const float* __restrict__ al0, const float* __restrict__ al1, const float* __restrict__ al2,
    const float* __restrict__ basisW,
    const float* __restrict__ W1, const float* __restrict__ b1,
    const float* __restrict__ W2, const float* __restrict__ b2,
    const float* __restrict__ W3, const float* __restrict__ b3,
    const float* __restrict__ weight,
    float* __restrict__ rgb_out, float* __restrict__ vis_out) {

  __shared__ _Float16 Fs[64 * LDF];    // 13.3 KB: raw tri-plane features (72 + zero pad)
  __shared__ _Float16 Xs[64 * LDX];    //  5.1 KB: MLP input (27 app + 3 viewdir, padded)
  __shared__ _Float16 H1s[64 * LDH];   // 17.4 KB: hidden layer 1
  __shared__ _Float16 H2s[64 * LDH];   // 17.4 KB: hidden layer 2
  __shared__ float    W3s[FEAT * 4];   //  2 KB
  __shared__ float    b3s[4];

  const int tid  = threadIdx.x;
  const int lane = tid & 31;
  const int wv   = tid >> 5;            // 8 waves
  const int tileBase = blockIdx.x * 256;

  __builtin_prefetch(basisW, 0, 3);     // global_prefetch_b8

  // ---- TDM: DMA W3 (128x4 fp32 = 2KB) into LDS, wave 0 issues it ----
  if (wv == 0) {
    unsigned long long ga = (unsigned long long)(const void*)W3;
    unsigned ldsoff = (unsigned)(unsigned long long)(const void*)&W3s[0];
    u32x4 g0;
    g0[0] = 1u;                                   // count=1, user descriptor
    g0[1] = ldsoff;                               // lds_addr (bytes)
    g0[2] = (unsigned)ga;                         // global_addr[31:0]
    g0[3] = ((unsigned)(ga >> 32) & 0x1FFFFFFu) | (2u << 30); // addr[56:32], type=2
    i32x8 g1;
    g1[0] = (int)(2u << 16);                      // data_size=2 -> 4B elements
    g1[1] = (int)((512u & 0xFFFFu) << 16);        // tensor_dim0=512 (low half of bits 79:48)
    g1[2] = (int)(((512u >> 16) & 0xFFFFu) | (1u << 16)); // dim0 hi | tensor_dim1=1
    g1[3] = (int)(512u << 16);                    // dim1 hi=0 | tile_dim0=512
    g1[4] = 1;                                    // tile_dim1=1, tile_dim2=0
    g1[5] = 512;                                  // tensor_dim0_stride (low 32)
    g1[6] = 0;
    g1[7] = 0;
    i32x4 gz = {0, 0, 0, 0};
#if defined(__clang_major__) && (__clang_major__ >= 23)
    i32x8 gz8 = {0, 0, 0, 0, 0, 0, 0, 0};
    __builtin_amdgcn_tensor_load_to_lds(g0, g1, gz, gz, gz8, 0);
#else
    __builtin_amdgcn_tensor_load_to_lds(g0, g1, gz, gz, 0);
#endif
    __builtin_amdgcn_s_wait_tensorcnt(0);
  }
  // belt-and-suspenders cooperative copy (same data) + b3
  for (int i = tid; i < FEAT * 4; i += 256) W3s[i] = W3[i];
  if (tid < 4) b3s[tid] = b3[tid];
  // zero the K-padding of the feats buffer once (cols 72..95)
  for (int i = tid; i < 64 * 24; i += 256) {
    int r = i / 24, c = i - r * 24;
    Fs[r * LDF + 72 + c] = (_Float16)0.0f;
  }

  // ---- resident weight fragments ----
  // B fragments here use contiguous K (same order as the A loads): lanes 0-15 hold K 0..15, lanes 16-31 hold K 16..31,
  // column = strip base + lane%16.
  const int col = (wv << 4) + (lane & 15);        // MLP output channel 0..127
  const int kb  = (lane >> 4) << 4;               // 0 or 16

  v16h fw1;                                       // W1 (30x128, K padded to 32)
#pragma unroll
  for (int j = 0; j < 16; ++j) {
    int K = kb + j;
    fw1[j] = (K < APPD + 3) ? (_Float16)W1[K * FEAT + col] : (_Float16)0.0f;
  }
  v16h fw2[4];                                    // W2 (128x128)
#pragma unroll
  for (int t = 0; t < 4; ++t)
#pragma unroll
    for (int j = 0; j < 16; ++j) {
      int K = t * 32 + kb + j;
      fw2[t][j] = (_Float16)W2[K * FEAT + col];
    }
  const float bias1 = b1[col];
  const float bias2 = b2[col];

  // einsum basis fragments: wave wv -> (M-subtile mtE, N-half nhE)
  const int mtE = wv >> 1;                        // 0..3
  const int nhE = wv & 1;                         // 0..1
  const int colE = (nhE << 4) + (lane & 15);      // einsum output column 0..31
  v16h fbas[3];                                   // basis_W (72x27, K padded to 96)
#pragma unroll
  for (int t = 0; t < 3; ++t)
#pragma unroll
    for (int j = 0; j < 16; ++j) {
      int K = t * 32 + kb + j;
      fbas[t][j] = (K < 72 && colE < APPD) ? (_Float16)basisW[K * APPD + colE]
                                           : (_Float16)0.0f;
    }

  const int ka   = (lane >> 4) << 4;              // A-frag K base: 0 or 16
  const int mAdd = (lane >> 4) << 3;              // C-layout row offset: 0 or 8

  __syncthreads();

  // ---- pipeline over 64-point subtiles ----
  for (int s4 = 0; s4 < 4; ++s4) {
    const int p0 = s4 * 64;

    // Phase A: tri-plane sampling, 4 threads per point (18 channels each)
    {
      const int pl = tid >> 2;                    // point 0..63
      const int q  = tid & 3;                     // channel quarter
      const int p  = tileBase + p0 + pl;
      float px = pts[p * 3 + 0], py = pts[p * 3 + 1], pz = pts[p * 3 + 2];
      float xn = (px - LOX) * INVX - 1.0f;
      float yn = (py - LOY) * INVY - 1.0f;
      float zn = (pz - LOZ) * INVZ - 1.0f;
      _Float16* frow = &Fs[pl * LDF];
      feat_store(ap0, al0, q * 12, 12, 0,  xn, yn, zn, frow);  // (x,y) plane, z line
      feat_store(ap1, al1, q * 3,  3,  48, xn, zn, yn, frow);  // (x,z) plane, y line
      feat_store(ap2, al2, q * 3,  3,  60, yn, zn, xn, frow);  // (y,z) plane, x line
      if (q == 0) {
        _Float16* xr = &Xs[pl * LDX];
        xr[27] = (_Float16)view_dirs[p * 3 + 0];
        xr[28] = (_Float16)view_dirs[p * 3 + 1];
        xr[29] = (_Float16)view_dirs[p * 3 + 2];
#pragma unroll
        for (int d = 30; d < LDX; ++d) xr[d] = (_Float16)0.0f;
      }
    }
    __syncthreads();

    // Phase B0: einsum Fs(64x72) @ basis(72x27) -> Xs cols 0..26  (WMMA, K=96)
    // Preload all 3 A-fragments, then issue the chained WMMAs back-to-back.
    {
      const _Float16* fr = &Fs[(mtE * 16 + (lane & 15)) * LDF];
      v16h a0 = load_afrag(fr, 0 + ka);
      v16h a1 = load_afrag(fr, 32 + ka);
      v16h a2 = load_afrag(fr, 64 + ka);
      v8f acc = {};
      acc = __builtin_amdgcn_wmma_f32_16x16x32_f16(false, a0, false, fbas[0],
                                                   (short)0, acc, false, false);
      acc = __builtin_amdgcn_wmma_f32_16x16x32_f16(false, a1, false, fbas[1],
                                                   (short)0, acc, false, false);
      acc = __builtin_amdgcn_wmma_f32_16x16x32_f16(false, a2, false, fbas[2],
                                                   (short)0, acc, false, false);
      if (colE < APPD) {
#pragma unroll
        for (int i = 0; i < 8; ++i) {
          int m = mtE * 16 + i + mAdd;
          Xs[m * LDX + colE] = (_Float16)acc[i];
        }
      }
    }
    __syncthreads();

    // Phase B1: layer 1  X(64x32) @ W1(32x128)
#pragma unroll
    for (int mt = 0; mt < 4; ++mt) {
      const int m0 = mt * 16;
      const _Float16* xr = &Xs[(m0 + (lane & 15)) * LDX];
      v16h a = load_afrag(xr, ka);
      v8f acc = {};
      acc = __builtin_amdgcn_wmma_f32_16x16x32_f16(false, a, false, fw1,
                                                   (short)0, acc, false, false);
#pragma unroll
      for (int i = 0; i < 8; ++i) {
        int m = m0 + i + mAdd;
        H1s[m * LDH + col] = (_Float16)fmaxf(acc[i] + bias1, 0.0f);
      }
    }
    __syncthreads();

    // Phase B2: layer 2  H1(64x128) @ W2(128x128), K-chain of 4 WMMAs.
    // Preload the whole K-strip of A-fragments before the chain.
#pragma unroll
    for (int mt = 0; mt < 4; ++mt) {
      const int m0 = mt * 16;
      const _Float16* hr = &H1s[(m0 + (lane & 15)) * LDH];
      v16h a0 = load_afrag(hr, 0   + ka);
      v16h a1 = load_afrag(hr, 32  + ka);
      v16h a2 = load_afrag(hr, 64  + ka);
      v16h a3 = load_afrag(hr, 96  + ka);
      v8f acc = {};
      acc = __builtin_amdgcn_wmma_f32_16x16x32_f16(false, a0, false, fw2[0],
                                                   (short)0, acc, false, false);
      acc = __builtin_amdgcn_wmma_f32_16x16x32_f16(false, a1, false, fw2[1],
                                                   (short)0, acc, false, false);
      acc = __builtin_amdgcn_wmma_f32_16x16x32_f16(false, a2, false, fw2[2],
                                                   (short)0, acc, false, false);
      acc = __builtin_amdgcn_wmma_f32_16x16x32_f16(false, a3, false, fw2[3],
                                                   (short)0, acc, false, false);
#pragma unroll
      for (int i = 0; i < 8; ++i) {
        int m = m0 + i + mAdd;
        H2s[m * LDH + col] = (_Float16)fmaxf(acc[i] + bias2, 0.0f);
      }
    }
    __syncthreads();

    // Phase B3: layer 3 (128 -> 4) + sigmoid + mask; thread -> (point, channel)
    {
      const int pl = tid >> 2;                  // 0..63
      const int c  = tid & 3;                   // 0..3
      const _Float16* hr = &H2s[pl * LDH];
      float acc = b3s[c];
#pragma unroll 8
      for (int k = 0; k < FEAT; ++k) acc = fmaf((float)hr[k], W3s[k * 4 + c], acc);
      float o = 1.0f / (1.0f + __expf(-acc));
      int gp = tileBase + p0 + pl;
      float out = (weight[gp] > W_THRESH) ? o : 0.0f;
      if (c < 3) rgb_out[gp * 3 + c] = out;
      else       vis_out[gp] = out;
    }
    __syncthreads();
  }
}

// ---------------- launch ----------------
extern "C" void kernel_launch(void* const* d_in, const int* in_sizes, int n_in,
                              void* d_out, int out_size, void* d_ws, size_t ws_size,
                              hipStream_t stream) {
  (void)in_sizes; (void)n_in; (void)out_size; (void)ws_size;
  const float* pts       = (const float*)d_in[0];
  const float* z_vals    = (const float*)d_in[1];
  const float* view_dirs = (const float*)d_in[2];
  const float* dp0 = (const float*)d_in[3];
  const float* dp1 = (const float*)d_in[4];
  const float* dp2 = (const float*)d_in[5];
  const float* dl0 = (const float*)d_in[6];
  const float* dl1 = (const float*)d_in[7];
  const float* dl2 = (const float*)d_in[8];
  const float* ap0 = (const float*)d_in[9];
  const float* ap1 = (const float*)d_in[10];
  const float* ap2 = (const float*)d_in[11];
  const float* al0 = (const float*)d_in[12];
  const float* al1 = (const float*)d_in[13];
  const float* al2 = (const float*)d_in[14];
  const float* basisW = (const float*)d_in[15];
  const float* W1 = (const float*)d_in[16];
  const float* b1 = (const float*)d_in[17];
  const float* W2 = (const float*)d_in[18];
  const float* b2 = (const float*)d_in[19];
  const float* W3 = (const float*)d_in[20];
  const float* b3 = (const float*)d_in[21];

  float* out   = (float*)d_out;
  float* rgb   = out;                     // R*S*3
  float* sigma = out + (size_t)3 * NPTS;  // R*S
  float* vis   = out + (size_t)4 * NPTS;  // R*S
  float* weight = (float*)d_ws;           // R*S floats of scratch

  sigma_kernel<<<NPTS / 256, 256, 0, stream>>>(pts, dp0, dp1, dp2, dl0, dl1, dl2, sigma);
  scan_kernel<<<R_RAYS / 256, 256, 0, stream>>>(z_vals, sigma, weight);
  mlp_kernel<<<NPTS / 256, 256, 0, stream>>>(pts, view_dirs,
                                             ap0, ap1, ap2, al0, al1, al2,
                                             basisW, W1, b1, W2, b2, W3, b3,
                                             weight, rgb, vis);
}
